// FineTuneModel_18614388261503
// MI455X (gfx1250) — compile-verified, matrix operand order checked by reading, not run on hardware
//
#include <hip/hip_runtime.h>
#include <hip/hip_bf16.h>
#include <stdint.h>
#include <math.h>

typedef __attribute__((ext_vector_type(16))) _Float16 v16h;
typedef __attribute__((ext_vector_type(8)))  _Float16 v8h;
typedef __attribute__((ext_vector_type(8)))  float    v8f;

#define NN 10000
#define NE 320000
#define NBH 96
#define NF 32
#define NHID 64
#define NTILES (NN / 16)          // 625, exact
#define TOTE ((long)NBH * NN * NF)
#define LDSB_STRIDE 40            // halves per column; 80B -> 16B-aligned b128 reads, spread banks

// ---------------- graph-constant precomputes ----------------

__global__ void k_init_deg(float* deg) {
    int i = blockIdx.x * blockDim.x + threadIdx.x;
    if (i < NN) deg[i] = 1.0f;                      // self loop
}

__global__ void k_degree(const int* __restrict__ ei, float* deg) {
    int e = blockIdx.x * blockDim.x + threadIdx.x;
    if (e < NE) atomicAdd(&deg[ei[NE + e]], 1.0f);  // dst row
}

__global__ void k_dinv(const float* __restrict__ deg, float* dinv) {
    int i = blockIdx.x * blockDim.x + threadIdx.x;
    if (i < NN) dinv[i] = rsqrtf(deg[i]);           // deg >= 1 always
}

__global__ void k_s1_init(const float* __restrict__ dinv, float* s1) {
    int i = blockIdx.x * blockDim.x + threadIdx.x;
    if (i < NN) s1[i] = dinv[i] * dinv[i];          // self-loop term of S*1
}

__global__ void k_s1_edges(const int* __restrict__ ei,
                           const float* __restrict__ dinv, float* s1) {
    int e = blockIdx.x * blockDim.x + threadIdx.x;
    if (e < NE) {
        int s = ei[e], d = ei[NE + e];
        atomicAdd(&s1[d], dinv[s] * dinv[d]);
    }
}

__global__ void k_s2_init(const float* __restrict__ dinv,
                          const float* __restrict__ s1, float* s2) {
    int i = blockIdx.x * blockDim.x + threadIdx.x;
    if (i < NN) s2[i] = dinv[i] * dinv[i] * s1[i];
}

__global__ void k_s2_edges(const int* __restrict__ ei,
                           const float* __restrict__ dinv,
                           const float* __restrict__ s1, float* s2) {
    int e = blockIdx.x * blockDim.x + threadIdx.x;
    if (e < NE) {
        int s = ei[e], d = ei[NE + e];
        atomicAdd(&s2[d], dinv[s] * dinv[d] * s1[s]);
    }
}

// Fold W1,W2,b1 into: W12 = W1[:32]@W2 (f16 for WMMA B), wlast = W1[32]@W2, bv = b1@W2
__global__ void k_weights(const float* __restrict__ W1, const float* __restrict__ b1,
                          const float* __restrict__ W2,
                          _Float16* __restrict__ W12h, float* __restrict__ wlast,
                          float* __restrict__ bv) {
    int tid = threadIdx.x;              // 1024 threads
    int i = tid >> 5, j = tid & 31;
    float acc = 0.0f;
    for (int k = 0; k < NHID; ++k) acc += W1[i * NHID + k] * W2[k * NF + j];
    W12h[i * NF + j] = (_Float16)acc;
    if (i == 0) {
        float a2 = 0.0f, a3 = 0.0f;
        for (int k = 0; k < NHID; ++k) {
            a2 += W1[32 * NHID + k] * W2[k * NF + j];
            a3 += b1[k] * W2[k * NF + j];
        }
        wlast[j] = a2;
        bv[j] = a3;
    }
}

// ---------------- per-pair sparse aggregation (S applied to 32-wide features) ----------------

// out = dinv[node]^2 * in, float4-vectorized (b128 load/store); fully initializes out (self-loop term)
__global__ void k_scale_copy4(const float* __restrict__ dinv,
                              const float4* __restrict__ in, float4* __restrict__ out) {
    long idx = (long)blockIdx.x * blockDim.x + threadIdx.x;
    if (idx < TOTE / 4) {
        int node = (int)((idx >> 3) % NN);      // 8 float4 per 32-feature row
        float w = dinv[node];
        w *= w;
        float4 v = in[idx];
        out[idx] = make_float4(w * v.x, w * v.y, w * v.z, w * v.w);
    }
}

// out[bh,dst,:] += dinv[src]*dinv[dst] * in[bh,src,:]
// block: 128 threads = 4 edges x 32 features (coalesced 128B read per edge,
// contiguous-cacheline float atomics on the 128B dst row)
__global__ void k_agg(const int* __restrict__ ei, const float* __restrict__ dinv,
                      const float* __restrict__ in, float* __restrict__ out) {
    int f  = threadIdx.x & 31;
    int le = threadIdx.x >> 5;
    long e = (long)blockIdx.x * 4 + le;
    if (e >= NE) return;
    int s = ei[e], d = ei[NE + e];
    float w = dinv[s] * dinv[d];
    long bh = blockIdx.y;
    float v = w * in[(bh * NN + s) * NF + f];
    atomicAdd(&out[(bh * NN + d) * NF + f], v);
}

// ---------------- WMMA epilogue: z <- tanh(z @ W12 + m*s2*wlast + s1*bv + b2) ----------------
// One wave per 16-node tile. A = z[16x32] f32->f16 via 4x global b128 per lane;
// B fragments built from LDS-staged transposed W12 via ds_load_b128.

__global__ void __launch_bounds__(256)
k_final(float* __restrict__ z,
        const float* __restrict__ s1, const float* __restrict__ s2,
        const _Float16* __restrict__ W12h,
        const float* __restrict__ wlast, const float* __restrict__ bv,
        const float* __restrict__ b2, const float* __restrict__ mask) {
    __shared__ _Float16 ldsB[32 * LDSB_STRIDE];   // W12 transposed: ldsB[n][k]

    // cooperative transposed stage of W12 (1024 halves) by 256 threads
    for (int t = threadIdx.x; t < 1024; t += 256) {
        int k = t >> 5, n = t & 31;
        ldsB[n * LDSB_STRIDE + k] = W12h[t];
    }
    __syncthreads();

    int wave = threadIdx.x >> 5;
    int lane = threadIdx.x & 31;
    int tile = blockIdx.x * 8 + wave;
    int bh   = blockIdx.y;
    if (tile >= NTILES) return;          // wave-uniform branch: EXEC all-1s for WMMA

    float mval = mask[bh];
    int hi   = lane >> 4;                // selects K half-group per ISA 16-bit A layout
    int mrow = lane & 15;                // A: M row;  B/C/D: N column
    int n    = lane & 15;
    int base = tile * 16;
    long rowbase = (long)bh * NN + base;

    // ---- A fragment: lane needs z[row][8hi..8hi+7] and z[row][16+8hi..+7] ----
    const float4* zrow = (const float4*)&z[(rowbase + mrow) * NF];
    float4 p0 = zrow[2 * hi];
    float4 p1 = zrow[2 * hi + 1];
    float4 p2 = zrow[4 + 2 * hi];
    float4 p3 = zrow[5 + 2 * hi];
    v16h a;
    a[0]  = (_Float16)p0.x; a[1]  = (_Float16)p0.y; a[2]  = (_Float16)p0.z; a[3]  = (_Float16)p0.w;
    a[4]  = (_Float16)p1.x; a[5]  = (_Float16)p1.y; a[6]  = (_Float16)p1.z; a[7]  = (_Float16)p1.w;
    a[8]  = (_Float16)p2.x; a[9]  = (_Float16)p2.y; a[10] = (_Float16)p2.z; a[11] = (_Float16)p2.w;
    a[12] = (_Float16)p3.x; a[13] = (_Float16)p3.y; a[14] = (_Float16)p3.z; a[15] = (_Float16)p3.w;

    // ---- B fragments from LDS (column n / n+16, same K grouping as A) ----
    v8h q0 = *(const v8h*)&ldsB[n * LDSB_STRIDE + 8 * hi];
    v8h q1 = *(const v8h*)&ldsB[n * LDSB_STRIDE + 16 + 8 * hi];
    v8h q2 = *(const v8h*)&ldsB[(n + 16) * LDSB_STRIDE + 8 * hi];
    v8h q3 = *(const v8h*)&ldsB[(n + 16) * LDSB_STRIDE + 16 + 8 * hi];
    v16h b0, b1f;
    #pragma unroll
    for (int i = 0; i < 8; ++i) {
        b0[i]      = q0[i];
        b0[i + 8]  = q1[i];
        b1f[i]     = q2[i];
        b1f[i + 8] = q3[i];
    }

    v8f c0 = {}, c1 = {};
    c0 = __builtin_amdgcn_wmma_f32_16x16x32_f16(false, a, false, b0,  (short)0, c0, false, false);
    c1 = __builtin_amdgcn_wmma_f32_16x16x32_f16(false, a, false, b1f, (short)0, c1, false, false);

    #pragma unroll
    for (int r = 0; r < 8; ++r) {
        int  m    = r + (hi ? 8 : 0);                      // C/D layout: VGPR r -> M=r / r+8
        long node = base + m;
        long gi   = ((long)bh * NN + node) * NF;
        float r1  = mval * s2[node];
        float r2  = s1[node];
        z[gi + n]      = tanhf(c0[r] + r1 * wlast[n]      + r2 * bv[n]      + b2[n]);
        z[gi + n + 16] = tanhf(c1[r] + r1 * wlast[n + 16] + r2 * bv[n + 16] + b2[n + 16]);
    }
}

// ---------------- host side ----------------

extern "C" void kernel_launch(void* const* d_in, const int* in_sizes, int n_in,
                              void* d_out, int out_size, void* d_ws, size_t ws_size,
                              hipStream_t stream) {
    const float* x    = (const float*)d_in[0];   // [4,24,10000,32]
    const float* mask = (const float*)d_in[1];   // [4,24]
    const int*   ei   = (const int*)d_in[2];     // [2,320000]
    const float* W1   = (const float*)d_in[3];   // [33,64]
    const float* b1   = (const float*)d_in[4];   // [64]
    const float* W2   = (const float*)d_in[5];   // [64,32]
    const float* b2   = (const float*)d_in[6];   // [32]
    float* out = (float*)d_out;                  // [4,24,10000,32], doubles as z buffer

    char* p = (char*)d_ws;
    float* deg   = (float*)p; p += NN * sizeof(float);
    float* dinv  = (float*)p; p += NN * sizeof(float);
    float* s1    = (float*)p; p += NN * sizeof(float);
    float* s2    = (float*)p; p += NN * sizeof(float);
    float* wlast = (float*)p; p += 32 * sizeof(float);
    float* bv    = (float*)p; p += 32 * sizeof(float);
    _Float16* W12h = (_Float16*)p; p += 32 * 32 * sizeof(_Float16);
    p = (char*)(((uintptr_t)p + 255) & ~(uintptr_t)255);
    float* y = (float*)p;                        // [96,10000,32] staging: y = S*x

    int gn = (NN + 255) / 256;
    int ge = (NE + 255) / 256;
    int g4 = (int)((TOTE / 4 + 255) / 256);

    // Graph-constant precomputes (cheap; recomputed every call for determinism)
    k_init_deg<<<gn, 256, 0, stream>>>(deg);
    k_degree  <<<ge, 256, 0, stream>>>(ei, deg);
    k_dinv    <<<gn, 256, 0, stream>>>(deg, dinv);
    k_s1_init <<<gn, 256, 0, stream>>>(dinv, s1);
    k_s1_edges<<<ge, 256, 0, stream>>>(ei, dinv, s1);
    k_s2_init <<<gn, 256, 0, stream>>>(dinv, s1, s2);
    k_s2_edges<<<ge, 256, 0, stream>>>(ei, dinv, s1, s2);
    k_weights <<<1, 1024, 0, stream>>>(W1, b1, W2, W12h, wlast, bv);

    // y = S * x   (self-loop init then edge scatter)
    k_scale_copy4<<<g4, 256, 0, stream>>>(dinv, (const float4*)x, (float4*)y);
    dim3 gagg((NE + 3) / 4, NBH);
    k_agg<<<gagg, 128, 0, stream>>>(ei, dinv, x, y);

    // z = S * y   (into d_out)
    k_scale_copy4<<<g4, 256, 0, stream>>>(dinv, (const float4*)y, (float4*)out);
    k_agg<<<gagg, 128, 0, stream>>>(ei, dinv, y, out);

    // out = tanh(z @ W12 + m*s2*wlast + s1*bv + b2), in place via WMMA
    dim3 gfin((NTILES + 7) / 8, NBH);
    k_final<<<gfin, 256, 0, stream>>>(out, s1, s2, W12h, wlast, bv, b2, mask);
}
